// EagerMultiheadAttention_21766894256719
// MI455X (gfx1250) — matrix operand order repaired and checked by reading, not compiled, not run on hardware
//
#include <hip/hip_runtime.h>
#include <math.h>

// ---------------------------------------------------------------------------
// EagerMultiheadAttention forward for MI455X (gfx1250, wave32, WMMA + TDM).
// GEMMs: v_wmma_f32_16x16x32_bf16, fp32 accumulate.
// Tile staging: tensor_load_to_lds (TDM, TENSORcnt) with double buffering.
// ---------------------------------------------------------------------------

typedef __bf16 bf16;
typedef __attribute__((ext_vector_type(16))) bf16 v16bf;
typedef __attribute__((ext_vector_type(8)))  float v8f;
typedef __attribute__((ext_vector_type(4)))  unsigned int u32x4;
typedef __attribute__((ext_vector_type(4)))  int i32x4;
typedef __attribute__((ext_vector_type(8)))  int i32x8;

#define EDIM   2048
#define E3DIM  6144
#define NHEAD  16
#define HDIM   128
#define BSZ    2
#define SEQ    2048
#define SCALE_QK 0.08838834764831845f  // 1/sqrt(128)

#if defined(__HIP_DEVICE_COMPILE__) && __has_builtin(__builtin_amdgcn_tensor_load_to_lds)
#define USE_TDM 1
#else
#define USE_TDM 0
#endif

static __device__ __forceinline__ v8f wmma_bf16f32(v16bf a, v16bf b, v8f c) {
    // 8 args: (neg_a, A, neg_b, B, c_mod, C, reuse_a, reuse_b)
    return __builtin_amdgcn_wmma_f32_16x16x32_bf16(false, a, false, b,
                                                   (short)0, c, false, false);
}

#if USE_TDM
// ---------------------------------------------------------------------------
// TDM: DMA a 2D bf16 tile (tile_w x tile_h, row stride stride_elems) from
// global memory into LDS at lds_addr, inserting pad_amount DWORDs of padding
// after every pad_interval DWORDs (codes per ISA: interval 3=16dw, 5=64dw;
// amount 3 = 4 dwords) so rows land on the padded LDS pitch.
// D# per CDNA5 ISA ch.8: group0 = {count=1, lds_addr, global_addr[56:0],
// type=2}; group1 = {data_size=2B, pad ctl, tensor dims, tile dims, strides}.
// ---------------------------------------------------------------------------
static __device__ __forceinline__ void tdm_load_tile_2d(
    const bf16* gaddr, unsigned lds_addr,
    int tile_w, int tile_h, int stride_elems,
    int pad_interval_code, int pad_amount_code)
{
    unsigned long long ga = (unsigned long long)(uintptr_t)gaddr;
    u32x4 g0;
    g0[0] = 1u;                                      // count=1, user D#
    g0[1] = lds_addr;                                // LDS byte address
    g0[2] = (unsigned)(ga & 0xFFFFFFFFu);            // global_addr[31:0]
    g0[3] = (unsigned)((ga >> 32) & 0x01FFFFFFu)     // global_addr[56:32]
          | (2u << 30);                              // type=2 ("image")

    unsigned td0 = (unsigned)stride_elems;           // tensor_dim0 (elems)
    unsigned td1 = (unsigned)tile_h;                 // tensor_dim1 (rows)
    unsigned long long s0 = (unsigned long long)stride_elems;

    i32x8 g1;
    g1[0] = (1 << 16)                                // data_size = 2 bytes
          | (1 << 20)                                // pad_enable
          | (pad_interval_code << 22)
          | (pad_amount_code << 25);                 // workgroup_mask = 0
    g1[1] = (int)((td0 & 0xFFFFu) << 16);            // tensor_dim0[15:0]
    g1[2] = (int)((td0 >> 16) | ((td1 & 0xFFFFu) << 16));
    g1[3] = (int)((td1 >> 16) | ((unsigned)tile_w << 16));   // tile_dim0
    g1[4] = (int)((unsigned)tile_h);                 // tile_dim1; tile_dim2=0
    g1[5] = (int)(s0 & 0xFFFFFFFFu);                 // tensor_dim0_stride lo
    g1[6] = (int)((s0 >> 32) & 0xFFFFu);             // stride hi; dim1 stride=0
    g1[7] = 0;

    i32x4 z4 = {0, 0, 0, 0};
#if defined(__clang_major__) && __clang_major__ >= 23
    i32x8 z8 = {0, 0, 0, 0, 0, 0, 0, 0};
    __builtin_amdgcn_tensor_load_to_lds(g0, g1, z4, z4, z8, 0);
#else
    __builtin_amdgcn_tensor_load_to_lds(g0, g1, z4, z4, 0);
#endif
}
#endif  // USE_TDM

// ---------------------------------------------------------------------------
// fp32 -> bf16 conversion
// ---------------------------------------------------------------------------
__global__ __launch_bounds__(256) void f32_to_bf16_kernel(
    const float* __restrict__ src, bf16* __restrict__ dst, int n)
{
    int i = blockIdx.x * 256 + threadIdx.x;
    if (i < n) dst[i] = (bf16)src[i];
}

// ---------------------------------------------------------------------------
// NT GEMM: C[M,N] = A[M,K] * B[N,K]^T + bias[N]
// A, B bf16 row-major, K contiguous. grid = (N/128, M/128), 256 threads.
// Double-buffered LDS tiles; TDM prefetches tile kt+1 while computing kt.
// ---------------------------------------------------------------------------
template <bool BF16OUT>
__global__ __launch_bounds__(256) void gemm_nt_kernel(
    const bf16* __restrict__ A, int lda,
    const bf16* __restrict__ B, int ldb,
    const float* __restrict__ bias,
    void* __restrict__ C, int ldc, int K)
{
    __shared__ bf16 sA[2][128][40];  // 32 K + 8 pad (row = 80B, 16B aligned)
    __shared__ bf16 sB[2][128][40];

    const int tid    = threadIdx.x;
    const int lane   = tid & 31;
    const int wid    = tid >> 5;
    const int lane15 = lane & 15;
    const int hi     = lane >> 4;          // 0: lanes 0-15, 1: lanes 16-31
    const int mBase  = blockIdx.y * 128;
    const int nBase  = blockIdx.x * 128;
    const int waveM  = wid & 1;
    const int waveN  = wid >> 1;

    v8f acc[4][2];
    const v8f vzero = {0.f,0.f,0.f,0.f,0.f,0.f,0.f,0.f};
#pragma unroll
    for (int mf = 0; mf < 4; ++mf)
#pragma unroll
        for (int nf = 0; nf < 2; ++nf) acc[mf][nf] = vzero;

    const int kSteps = K >> 5;

    // stage tile 0 into buffer 0
#if USE_TDM
    if (wid == 0) {
        tdm_load_tile_2d(A + (size_t)mBase * lda,
                         (unsigned)(uintptr_t)&sA[0][0][0], 32, 128, lda, 3, 3);
        tdm_load_tile_2d(B + (size_t)nBase * ldb,
                         (unsigned)(uintptr_t)&sB[0][0][0], 32, 128, ldb, 3, 3);
    }
#else
#pragma unroll
    for (int c = tid; c < 512; c += 256) {
        int row = c >> 2, col8 = (c & 3) * 8;
        *(uint4*)&sA[0][row][col8] =
            *(const uint4*)(A + (size_t)(mBase + row) * lda + col8);
        *(uint4*)&sB[0][row][col8] =
            *(const uint4*)(B + (size_t)(nBase + row) * ldb + col8);
    }
#endif

    for (int kt = 0; kt < kSteps; ++kt) {
        const int cur = kt & 1;
#if USE_TDM
        if (wid == 0) __builtin_amdgcn_s_wait_tensorcnt(0);
#endif
        __syncthreads();

        if (kt + 1 < kSteps) {
            const int nxt = (kt + 1) & 1;
            const int k1  = (kt + 1) * 32;
#if USE_TDM
            if (wid == 0) {
                tdm_load_tile_2d(A + (size_t)mBase * lda + k1,
                                 (unsigned)(uintptr_t)&sA[nxt][0][0],
                                 32, 128, lda, 3, 3);
                tdm_load_tile_2d(B + (size_t)nBase * ldb + k1,
                                 (unsigned)(uintptr_t)&sB[nxt][0][0],
                                 32, 128, ldb, 3, 3);
            }
#else
#pragma unroll
            for (int c = tid; c < 512; c += 256) {
                int row = c >> 2, col8 = (c & 3) * 8;
                *(uint4*)&sA[nxt][row][col8] =
                    *(const uint4*)(A + (size_t)(mBase + row) * lda + k1 + col8);
                *(uint4*)&sB[nxt][row][col8] =
                    *(const uint4*)(B + (size_t)(nBase + row) * ldb + k1 + col8);
            }
#endif
        }

        v16bf bfrag[2];
#pragma unroll
        for (int nf = 0; nf < 2; ++nf) {
            int n = waveN * 32 + nf * 16 + lane15;
            *(uint4*)&bfrag[nf]         = *(uint4*)&sB[cur][n][hi * 8];
            *(((uint4*)&bfrag[nf]) + 1) = *(uint4*)&sB[cur][n][16 + hi * 8];
        }
#pragma unroll
        for (int mf = 0; mf < 4; ++mf) {
            int m = waveM * 64 + mf * 16 + lane15;
            v16bf a;
            *(uint4*)&a         = *(uint4*)&sA[cur][m][hi * 8];
            *(((uint4*)&a) + 1) = *(uint4*)&sA[cur][m][16 + hi * 8];
            acc[mf][0] = wmma_bf16f32(a, bfrag[0], acc[mf][0]);
            acc[mf][1] = wmma_bf16f32(a, bfrag[1], acc[mf][1]);
        }
        __syncthreads();
    }

    // C layout: VGPR r -> row M = r + 8*hi; col N = lane15
#pragma unroll
    for (int mf = 0; mf < 4; ++mf)
#pragma unroll
        for (int nf = 0; nf < 2; ++nf)
#pragma unroll
            for (int r = 0; r < 8; ++r) {
                int gm = mBase + waveM * 64 + mf * 16 + r + hi * 8;
                int gn = nBase + waveN * 32 + nf * 16 + lane15;
                float v = acc[mf][nf][r] + bias[gn];
                if constexpr (BF16OUT)
                    ((bf16*)C)[(size_t)gm * ldc + gn] = (bf16)v;
                else
                    ((float*)C)[(size_t)gm * ldc + gn] = v;
            }
}

// ---------------------------------------------------------------------------
// Fused scores + softmax per (b, h, 32-row block).
// S[32, 2048] (fp32) lives in dynamic LDS (256 KB <= 320 KB/WGP).
// grid = (SEQ/32, NHEAD, BSZ), 256 threads.
// ---------------------------------------------------------------------------
__global__ __launch_bounds__(256) void attn_scores_softmax_kernel(
    const bf16* __restrict__ qkv, bf16* __restrict__ attn)
{
    extern __shared__ char smem[];
    float* sS = (float*)smem;                                   // [32][2048]
    bf16*  sQ = (bf16*)(smem + 32 * 2048 * sizeof(float));      // [32][136]

    const int tid    = threadIdx.x;
    const int lane   = tid & 31;
    const int wid    = tid >> 5;
    const int lane15 = lane & 15;
    const int hi     = lane >> 4;
    const int rowBase = blockIdx.x * 32;
    const int hh      = blockIdx.y;
    const int bb      = blockIdx.z;

    // stage Q tile [32][128] bf16 (q section = cols 0..E-1 of qkv)
#pragma unroll
    for (int c = tid; c < 512; c += 256) {
        int row = c >> 4, col8 = (c & 15) * 8;
        const bf16* src = qkv + (size_t)(bb * SEQ + rowBase + row) * E3DIM
                              + hh * HDIM + col8;
        *(uint4*)&sQ[row * 136 + col8] = *(const uint4*)src;
    }
    __syncthreads();

    // scores: each wave covers 16 key tiles (16 keys each) of the 2048 keys
    for (int t = 0; t < 16; ++t) {
        const int nt = wid * 16 + t;
        const v8f vzero = {0.f,0.f,0.f,0.f,0.f,0.f,0.f,0.f};
        v8f acc0 = vzero, acc1 = vzero;
#pragma unroll
        for (int ks = 0; ks < 4; ++ks) {        // K = HDIM = 128 -> 4 x 32
            const int koff = ks * 32 + hi * 8;
            v16bf a0, a1;
            *(uint4*)&a0         = *(uint4*)&sQ[lane15 * 136 + koff];
            *(((uint4*)&a0) + 1) = *(uint4*)&sQ[lane15 * 136 + koff + 16];
            *(uint4*)&a1         = *(uint4*)&sQ[(16 + lane15) * 136 + koff];
            *(((uint4*)&a1) + 1) = *(uint4*)&sQ[(16 + lane15) * 136 + koff + 16];
            // B fragment: key row n, K = hi*8+0..7 and 16+hi*8+0..7 (NT form)
            const int n = nt * 16 + lane15;
            const bf16* kp = qkv + (size_t)(bb * SEQ + n) * E3DIM
                                 + EDIM + hh * HDIM + ks * 32 + hi * 8;
            v16bf bfr;
            *(uint4*)&bfr         = *(const uint4*)kp;
            *(((uint4*)&bfr) + 1) = *(const uint4*)(kp + 16);
            acc0 = wmma_bf16f32(a0, bfr, acc0);
            acc1 = wmma_bf16f32(a1, bfr, acc1);
        }
#pragma unroll
        for (int r = 0; r < 8; ++r) {
            sS[(r + hi * 8) * 2048 + nt * 16 + lane15]      = acc0[r] * SCALE_QK;
            sS[(16 + r + hi * 8) * 2048 + nt * 16 + lane15] = acc1[r] * SCALE_QK;
        }
    }
    __syncthreads();

    // softmax: 8 threads per row (rows 0..31), shuffle-reduce within 8 lanes
    const int row = tid >> 3;
    const int sub = tid & 7;
    float mx = -INFINITY;
    for (int c = sub; c < 2048; c += 8) mx = fmaxf(mx, sS[row * 2048 + c]);
#pragma unroll
    for (int o = 4; o >= 1; o >>= 1) mx = fmaxf(mx, __shfl_xor(mx, o, 8));
    float sum = 0.f;
    for (int c = sub; c < 2048; c += 8) {
        float e = __expf(sS[row * 2048 + c] - mx);
        sS[row * 2048 + c] = e;
        sum += e;
    }
#pragma unroll
    for (int o = 4; o >= 1; o >>= 1) sum += __shfl_xor(sum, o, 8);
    const float inv = 1.f / sum;
    bf16* dst = attn + ((size_t)(bb * NHEAD + hh) * SEQ + rowBase + row) * SEQ;
    for (int c = sub; c < 2048; c += 8)
        dst[c] = (bf16)(sS[row * 2048 + c] * inv);
}

// ---------------------------------------------------------------------------
// NN GEMM per (b,h): ctx[SEQ,128] = P[SEQ,SEQ] * V[SEQ,128]
// grid = (SEQ/128, NHEAD, BSZ), 256 threads; each wave: 16 rows x 128 cols.
// Double-buffered LDS; TDM prefetch of P and V tiles.
// ---------------------------------------------------------------------------
__global__ __launch_bounds__(256) void attn_ctx_kernel(
    const bf16* __restrict__ attn, const bf16* __restrict__ qkv,
    bf16* __restrict__ ctx)
{
    __shared__ bf16 sA[2][128][40];   // P tile: 128 rows x 32 K (pad 4 dw/16 dw)
    __shared__ bf16 sB[2][32][136];   // V tile: 32 K-rows x 128 N (pad 4 dw/64 dw)

    const int tid    = threadIdx.x;
    const int lane   = tid & 31;
    const int wid    = tid >> 5;
    const int lane15 = lane & 15;
    const int hi     = lane >> 4;
    const int mBase  = blockIdx.x * 128;
    const int hh     = blockIdx.y;
    const int bb     = blockIdx.z;

    const bf16* P = attn + (size_t)(bb * NHEAD + hh) * SEQ * SEQ;
    const bf16* V = qkv + (size_t)bb * SEQ * E3DIM + 2 * EDIM + hh * HDIM;

    v8f acc[8];
    const v8f vzero = {0.f,0.f,0.f,0.f,0.f,0.f,0.f,0.f};
#pragma unroll
    for (int nf = 0; nf < 8; ++nf) acc[nf] = vzero;

    const int kSteps = SEQ / 32;

#if USE_TDM
    if (wid == 0) {
        tdm_load_tile_2d(P + (size_t)mBase * SEQ,
                         (unsigned)(uintptr_t)&sA[0][0][0], 32, 128, SEQ, 3, 3);
        tdm_load_tile_2d(V, (unsigned)(uintptr_t)&sB[0][0][0],
                         128, 32, E3DIM, 5, 3);
    }
#else
#pragma unroll
    for (int c = tid; c < 512; c += 256) {
        int rowA = c >> 2, colA = (c & 3) * 8;
        *(uint4*)&sA[0][rowA][colA] =
            *(const uint4*)(P + (size_t)(mBase + rowA) * SEQ + colA);
        int rowB = c >> 4, colB = (c & 15) * 8;
        *(uint4*)&sB[0][rowB][colB] =
            *(const uint4*)(V + (size_t)rowB * E3DIM + colB);
    }
#endif

    for (int kt = 0; kt < kSteps; ++kt) {
        const int cur = kt & 1;
#if USE_TDM
        if (wid == 0) __builtin_amdgcn_s_wait_tensorcnt(0);
#endif
        __syncthreads();

        if (kt + 1 < kSteps) {
            const int nxt = (kt + 1) & 1;
            const int k1  = (kt + 1) * 32;
#if USE_TDM
            if (wid == 0) {
                tdm_load_tile_2d(P + (size_t)mBase * SEQ + k1,
                                 (unsigned)(uintptr_t)&sA[nxt][0][0],
                                 32, 128, SEQ, 3, 3);
                tdm_load_tile_2d(V + (size_t)k1 * E3DIM,
                                 (unsigned)(uintptr_t)&sB[nxt][0][0],
                                 128, 32, E3DIM, 5, 3);
            }
#else
#pragma unroll
            for (int c = tid; c < 512; c += 256) {
                int rowA = c >> 2, colA = (c & 3) * 8;
                *(uint4*)&sA[nxt][rowA][colA] =
                    *(const uint4*)(P + (size_t)(mBase + rowA) * SEQ + k1 + colA);
                int rowB = c >> 4, colB = (c & 15) * 8;
                *(uint4*)&sB[nxt][rowB][colB] =
                    *(const uint4*)(V + (size_t)(k1 + rowB) * E3DIM + colB);
            }
#endif
        }

        v16bf a;
        const int m = wid * 16 + lane15;
        *(uint4*)&a         = *(uint4*)&sA[cur][m][hi * 8];
        *(((uint4*)&a) + 1) = *(uint4*)&sA[cur][m][16 + hi * 8];
#pragma unroll
        for (int nf = 0; nf < 8; ++nf) {
            v16bf b;   // NN form: gather column n across 16 K rows
#pragma unroll
            for (int i = 0; i < 16; ++i)
                b[i] = sB[cur][(i < 8 ? hi * 8 + i : 8 + hi * 8 + i)][nf * 16 + lane15];
            acc[nf] = wmma_bf16f32(a, b, acc[nf]);
        }
        __syncthreads();
    }

#pragma unroll
    for (int nf = 0; nf < 8; ++nf)
#pragma unroll
        for (int r = 0; r < 8; ++r) {
            int gm = mBase + wid * 16 + r + hi * 8;
            int gn = nf * 16 + lane15;
            ctx[(size_t)(bb * SEQ + gm) * EDIM + hh * HDIM + gn] =
                (bf16)acc[nf][r];
        }
}

// ---------------------------------------------------------------------------
// attn_avg[b,q,k] = mean over heads of attn[b,h,q,k]
// ---------------------------------------------------------------------------
__global__ __launch_bounds__(256) void head_mean_kernel(
    const bf16* __restrict__ attn, float* __restrict__ out)
{
    size_t idx = (size_t)blockIdx.x * 256 + threadIdx.x;   // < BSZ*SEQ*SEQ
    size_t bb  = idx / ((size_t)SEQ * SEQ);
    size_t rem = idx % ((size_t)SEQ * SEQ);
    float s = 0.f;
#pragma unroll
    for (int h = 0; h < NHEAD; ++h)
        s += (float)attn[(bb * NHEAD + h) * (size_t)SEQ * SEQ + rem];
    out[idx] = s * (1.0f / NHEAD);
}

// ---------------------------------------------------------------------------
// Host launch
// ---------------------------------------------------------------------------
extern "C" void kernel_launch(void* const* d_in, const int* in_sizes, int n_in,
                              void* d_out, int out_size, void* d_ws, size_t ws_size,
                              hipStream_t stream)
{
    const float* q_f   = (const float*)d_in[0];  // query [B,L,E]
    const float* W_in  = (const float*)d_in[3];  // [3E,E]
    const float* b_in  = (const float*)d_in[4];  // [3E]
    const float* W_out = (const float*)d_in[5];  // [E,E]
    const float* b_out = (const float*)d_in[6];  // [E]

    char* ws = (char*)d_ws;
    bf16* qbf    = (bf16*)(ws);                       // 16 MB  query bf16
    bf16* winbf  = (bf16*)(ws + 16777216ull);         // 24 MB  W_in bf16
    bf16* woutbf = (bf16*)(ws + 41943040ull);         //  8 MB  W_out bf16
    bf16* qkvbf  = (bf16*)(ws + 50331648ull);         // 48 MB  qkv bf16
    bf16* attnbf = (bf16*)(ws + 100663296ull);        // 256 MB attn probs bf16
    bf16* ctxbf  = (bf16*)(ws + 369098752ull);        // 16 MB  ctx bf16

    // 1) fp32 -> bf16 conversions
    f32_to_bf16_kernel<<<(4096 * 2048) / 256, 256, 0, stream>>>(q_f,   qbf,    4096 * 2048);
    f32_to_bf16_kernel<<<(6144 * 2048) / 256, 256, 0, stream>>>(W_in,  winbf,  6144 * 2048);
    f32_to_bf16_kernel<<<(2048 * 2048) / 256, 256, 0, stream>>>(W_out, woutbf, 2048 * 2048);

    // 2) qkv = query * W_in^T + b_in   (M=4096, N=6144, K=2048) -> bf16
    gemm_nt_kernel<true><<<dim3(6144 / 128, 4096 / 128), 256, 0, stream>>>(
        qbf, EDIM, winbf, EDIM, b_in, (void*)qkvbf, E3DIM, EDIM);

    // 3) scores + softmax (dynamic LDS: 256 KB scores + 8.5 KB Q tile)
    const int smemBytes = 32 * 2048 * (int)sizeof(float) + 32 * 136 * (int)sizeof(bf16);
    hipFuncSetAttribute(reinterpret_cast<const void*>(attn_scores_softmax_kernel),
                        hipFuncAttributeMaxDynamicSharedMemorySize, smemBytes);
    attn_scores_softmax_kernel<<<dim3(SEQ / 32, NHEAD, BSZ), 256, smemBytes, stream>>>(
        qkvbf, attnbf);

    // 4) ctx = P * V per (b,h)
    attn_ctx_kernel<<<dim3(SEQ / 128, NHEAD, BSZ), 256, 0, stream>>>(
        attnbf, qkvbf, ctxbf);

    // 5) out = ctx * W_out^T + b_out -> fp32 d_out[0 : B*L*E]
    gemm_nt_kernel<false><<<dim3(2048 / 128, 4096 / 128), 256, 0, stream>>>(
        ctxbf, EDIM, woutbf, EDIM, b_out, d_out, EDIM, EDIM);

    // 6) attn_avg -> d_out[B*L*E : ]
    head_mean_kernel<<<(BSZ * SEQ * SEQ) / 256, 256, 0, stream>>>(
        attnbf, (float*)d_out + (size_t)BSZ * SEQ * EDIM);
}
